// Decoder_28527172780268
// MI455X (gfx1250) — matrix operand order repaired and checked by reading, not compiled, not run on hardware
//
#include <hip/hip_runtime.h>
#include <hip/hip_bf16.h>
#include <stdint.h>

// Shapes
#define Bsz 128
#define Ssz 1024
#define Esz 256
#define ENCsz 512
#define Hsz 512
#define Vsz 50257

typedef __attribute__((ext_vector_type(16))) __bf16 v16bf;
typedef __attribute__((ext_vector_type(8)))  float  v8f;

__device__ __forceinline__ uint32_t pack_bf16x2(float a, float b) {
  uint32_t ua = __builtin_bit_cast(uint32_t, a);
  uint32_t ub = __builtin_bit_cast(uint32_t, b);
  ua = (ua + 0x7FFFu + ((ua >> 16) & 1u)) >> 16;   // RNE f32 -> bf16
  ub = (ub + 0x7FFFu + ((ub >> 16) & 1u)) >> 16;
  return ua | (ub << 16);
}

__device__ __forceinline__ float sigmoidf_(float x) {
  return 1.0f / (1.0f + __expf(-x));
}

// ------- bf16 WMMA GEMM core: 128x64 tile, K-blocked 64, software-pipelined -------
#define BM 128
#define BN 64
#define BKT 64
#define KPAIRS (BKT / 2)          // 32 dword-pairs per row
#define LDSP 36                   // padded pair stride -> conflict-free ds_read
#define APL ((BM * KPAIRS) / 256) // 16 pairs/thread for A
#define BPL ((BN * KPAIRS) / 256) // 8  pairs/thread for B

union FragU { uint32_t u[8]; v16bf v; };

// Load one B fragment (same k order as A: lanes 0-15 K=0..15, lanes 16-31 K=16..31; 2 K/VGPR)
__device__ __forceinline__ void load_bfrag(FragU& bf, const uint32_t* Bs,
                                           int ni, int lane, int kk) {
  int n = ni * 16 + (lane & 15);
  int kbase = (lane >> 4) * 16;
  #pragma unroll
  for (int j = 0; j < 8; ++j)
    bf.u[j] = Bs[n * LDSP + (kk + kbase + 2 * j) / 2];
}

// Load one A fragment (lanes 0-15 K=0..15, lanes 16-31 K=16..31; 2 K/VGPR)
__device__ __forceinline__ void load_afrag(FragU& af, const uint32_t* As,
                                           int mi, int lane, int kk) {
  int m   = mi * 16 + (lane & 15);
  int kb2 = (lane >> 4) * 16;
  #pragma unroll
  for (int j = 0; j < 4; ++j)
    af.u[j] = As[m * LDSP + (kk + kb2 + 2 * j) / 2];
  #pragma unroll
  for (int j = 0; j < 4; ++j)
    af.u[4 + j] = As[m * LDSP + (kk + 8 + kb2 + 2 * j) / 2];
}

// tiles per wave: 4 M sub-tiles sharing one B fragment
// wave w: ni = w & 3, mi0 = (w >> 2) * 4

__global__ __launch_bounds__(256)
void gemm_bf16_wmma(const float* __restrict__ A, int lda,
                    const float* __restrict__ B, int ldb,
                    float* __restrict__ C, int ldc,
                    const float* __restrict__ bias,
                    int M, int N, int K, int accumulate)
{
  __shared__ uint32_t As[BM * LDSP];
  __shared__ uint32_t Bs[BN * LDSP];

  const int mb   = blockIdx.x * BM;
  const int nb   = blockIdx.y * BN;
  const int tid  = threadIdx.x;
  const int lane = tid & 31;
  const int wave = tid >> 5;
  const int ni   = wave & 3;
  const int mi0  = (wave >> 2) * 4;

  v8f acc[4] = {{}, {}, {}, {}};
  float2 ra[APL];
  float2 rb[BPL];

  // prologue: load K-block 0 into registers
  #pragma unroll
  for (int i = 0; i < APL; ++i) {
    int p = tid + i * 256, r = p >> 5, kp = p & 31;
    int gr = mb + r;
    ra[i] = make_float2(0.f, 0.f);
    if (gr < M) ra[i] = *(const float2*)(A + (size_t)gr * lda + kp * 2);
  }
  #pragma unroll
  for (int i = 0; i < BPL; ++i) {
    int p = tid + i * 256, r = p >> 5, kp = p & 31;
    int gr = nb + r;
    rb[i] = make_float2(0.f, 0.f);
    if (gr < N) rb[i] = *(const float2*)(B + (size_t)gr * ldb + kp * 2);
  }

  for (int kb = 0; kb < K; kb += BKT) {
    __syncthreads();   // previous compute done reading LDS
    #pragma unroll
    for (int i = 0; i < APL; ++i) {
      int p = tid + i * 256, r = p >> 5, kp = p & 31;
      As[r * LDSP + kp] = pack_bf16x2(ra[i].x, ra[i].y);
    }
    #pragma unroll
    for (int i = 0; i < BPL; ++i) {
      int p = tid + i * 256, r = p >> 5, kp = p & 31;
      Bs[r * LDSP + kp] = pack_bf16x2(rb[i].x, rb[i].y);
    }
    __syncthreads();

    // issue global loads for next block (latency hidden behind WMMAs below)
    const int kn = kb + BKT;
    if (kn < K) {
      #pragma unroll
      for (int i = 0; i < APL; ++i) {
        int p = tid + i * 256, r = p >> 5, kp = p & 31;
        int gr = mb + r;
        ra[i] = make_float2(0.f, 0.f);
        if (gr < M) ra[i] = *(const float2*)(A + (size_t)gr * lda + kn + kp * 2);
      }
      #pragma unroll
      for (int i = 0; i < BPL; ++i) {
        int p = tid + i * 256, r = p >> 5, kp = p & 31;
        int gr = nb + r;
        rb[i] = make_float2(0.f, 0.f);
        if (gr < N) rb[i] = *(const float2*)(B + (size_t)gr * ldb + kn + kp * 2);
      }
    }

    // compute: 2 kk steps; group all fragment loads, then 4 back-to-back WMMAs
    #pragma unroll
    for (int kk = 0; kk < BKT; kk += 32) {
      FragU bf, af[4];
      load_bfrag(bf, Bs, ni, lane, kk);
      #pragma unroll
      for (int t = 0; t < 4; ++t) load_afrag(af[t], As, mi0 + t, lane, kk);
      #pragma unroll
      for (int t = 0; t < 4; ++t)
        acc[t] = __builtin_amdgcn_wmma_f32_16x16x32_bf16(false, af[t].v, false, bf.v,
                                                         (short)0, acc[t], false, false);
    }
  }

  // epilogue (C/D layout: lane holds n = 16*ni+(lane&15); V r holds m = r + 8*(lane>>4))
  int ncol = nb + ni * 16 + (lane & 15);
  if (ncol < N) {
    float bv = bias ? bias[ncol] : 0.f;
    #pragma unroll
    for (int t = 0; t < 4; ++t) {
      #pragma unroll
      for (int r = 0; r < 8; ++r) {
        int mrow = mb + (mi0 + t) * 16 + r + 8 * (lane >> 4);
        if (mrow < M) {
          float* dst = C + (size_t)mrow * ldc + ncol;
          float val = acc[t][r] + bv;
          if (accumulate) val += *dst;
          *dst = val;
        }
      }
    }
  }
}

// ------------- Fused attention scores: scores[b,s] += sum_h v[h]*tanh(energy) -------------
// grid: (S/128, H/64, B). energy tile lives only in accumulators.
__global__ __launch_bounds__(256)
void attn_scores_wmma(const float* __restrict__ enc,      // [B,S,ENC]
                      const float* __restrict__ Wa,       // [H, H+ENC] (cols H..)
                      const float* __restrict__ ba,       // [H]
                      const float* __restrict__ vvec,     // [H]
                      const float* __restrict__ dec_part, // [B,H]
                      float* __restrict__ scores)         // [B,S], pre-zeroed
{
  __shared__ uint32_t As[BM * LDSP];
  __shared__ uint32_t Bs[BN * LDSP];
  __shared__ float sc[BM];

  const int s0 = blockIdx.x * BM;
  const int hb = blockIdx.y * BN;
  const int b  = blockIdx.z;
  const int tid  = threadIdx.x;
  const int lane = tid & 31;
  const int wave = tid >> 5;
  const int ni   = wave & 3;
  const int mi0  = (wave >> 2) * 4;

  if (tid < BM) sc[tid] = 0.f;

  const float* Abase = enc + ((size_t)b * Ssz + s0) * ENCsz;
  const float* Bbase = Wa + Hsz + (size_t)hb * (Hsz + ENCsz);
  const int ldb = Hsz + ENCsz;

  v8f acc[4] = {{}, {}, {}, {}};
  float2 ra[APL];
  float2 rb[BPL];

  #pragma unroll
  for (int i = 0; i < APL; ++i) {
    int p = tid + i * 256, r = p >> 5, kp = p & 31;
    ra[i] = *(const float2*)(Abase + (size_t)r * ENCsz + kp * 2);
  }
  #pragma unroll
  for (int i = 0; i < BPL; ++i) {
    int p = tid + i * 256, r = p >> 5, kp = p & 31;
    rb[i] = *(const float2*)(Bbase + (size_t)r * ldb + kp * 2);
  }

  for (int kb = 0; kb < ENCsz; kb += BKT) {
    __syncthreads();
    #pragma unroll
    for (int i = 0; i < APL; ++i) {
      int p = tid + i * 256, r = p >> 5, kp = p & 31;
      As[r * LDSP + kp] = pack_bf16x2(ra[i].x, ra[i].y);
    }
    #pragma unroll
    for (int i = 0; i < BPL; ++i) {
      int p = tid + i * 256, r = p >> 5, kp = p & 31;
      Bs[r * LDSP + kp] = pack_bf16x2(rb[i].x, rb[i].y);
    }
    __syncthreads();

    const int kn = kb + BKT;
    if (kn < ENCsz) {
      #pragma unroll
      for (int i = 0; i < APL; ++i) {
        int p = tid + i * 256, r = p >> 5, kp = p & 31;
        ra[i] = *(const float2*)(Abase + (size_t)r * ENCsz + kn + kp * 2);
      }
      #pragma unroll
      for (int i = 0; i < BPL; ++i) {
        int p = tid + i * 256, r = p >> 5, kp = p & 31;
        rb[i] = *(const float2*)(Bbase + (size_t)r * ldb + kn + kp * 2);
      }
    }

    #pragma unroll
    for (int kk = 0; kk < BKT; kk += 32) {
      FragU bf, af[4];
      load_bfrag(bf, Bs, ni, lane, kk);
      #pragma unroll
      for (int t = 0; t < 4; ++t) load_afrag(af[t], As, mi0 + t, lane, kk);
      #pragma unroll
      for (int t = 0; t < 4; ++t)
        acc[t] = __builtin_amdgcn_wmma_f32_16x16x32_bf16(false, af[t].v, false, bf.v,
                                                         (short)0, acc[t], false, false);
    }
  }

  // fused epilogue: tanh(energy + dec_part + ba) * v[h], reduce over h
  const int hcol = hb + ni * 16 + (lane & 15);
  const float dadd  = dec_part[(size_t)b * Hsz + hcol] + ba[hcol];
  const float scale = vvec[hcol];
  #pragma unroll
  for (int t = 0; t < 4; ++t) {
    #pragma unroll
    for (int r = 0; r < 8; ++r) {
      float val = tanhf(acc[t][r] + dadd) * scale;
      val += __shfl_xor(val, 1);
      val += __shfl_xor(val, 2);
      val += __shfl_xor(val, 4);
      val += __shfl_xor(val, 8);
      if ((lane & 15) == 0) {
        int m = (mi0 + t) * 16 + r + 8 * (lane >> 4);
        atomicAdd(&sc[m], val);
      }
    }
  }
  __syncthreads();
  if (tid < BM) atomicAdd(&scores[(size_t)b * Ssz + s0 + tid], sc[tid]);
}

// ---------------- embedding gather + zero scores ----------------
__global__ __launch_bounds__(256)
void embed_and_zero(const int* __restrict__ ids, const float* __restrict__ emb,
                    float* __restrict__ embedded, float* __restrict__ scores)
{
  int t = blockIdx.x * 256 + threadIdx.x;
  if (t < Bsz * Esz) {
    int b = t >> 8, e = t & 255;
    embedded[t] = emb[(size_t)ids[b] * Esz + e];
  }
  if (t < Bsz * Ssz) scores[t] = 0.f;
}

// ---------------- softmax over S (per b) + zero context ----------------
__global__ __launch_bounds__(256)
void softmax_zero_ctx(float* __restrict__ scores, float* __restrict__ context)
{
  const int b = blockIdx.x, tid = threadIdx.x;
  __shared__ float red[256];
  float* row = scores + (size_t)b * Ssz;
  float x[4];
  float m = -3.402823466e38f;
  #pragma unroll
  for (int i = 0; i < 4; ++i) { x[i] = row[tid + 256 * i]; m = fmaxf(m, x[i]); }
  red[tid] = m; __syncthreads();
  for (int s = 128; s > 0; s >>= 1) { if (tid < s) red[tid] = fmaxf(red[tid], red[tid + s]); __syncthreads(); }
  m = red[0]; __syncthreads();
  float sum = 0.f;
  #pragma unroll
  for (int i = 0; i < 4; ++i) { x[i] = __expf(x[i] - m); sum += x[i]; }
  red[tid] = sum; __syncthreads();
  for (int s = 128; s > 0; s >>= 1) { if (tid < s) red[tid] += red[tid + s]; __syncthreads(); }
  float inv = 1.f / red[0];
  #pragma unroll
  for (int i = 0; i < 4; ++i) row[tid + 256 * i] = x[i] * inv;
  context[(size_t)b * ENCsz + tid] = 0.f;
  context[(size_t)b * ENCsz + 256 + tid] = 0.f;
}

// ---------------- context[b,d] = sum_s attn[b,s]*enc[b,s,d]  (grid: S-chunks x B) --------
__global__ __launch_bounds__(256)
void context_accum(const float* __restrict__ enc, const float* __restrict__ attn,
                   float* __restrict__ context)
{
  const int b = blockIdx.y, sc0 = blockIdx.x * 128, tid = threadIdx.x;
  __shared__ float aw[128];
  if (tid < 128) aw[tid] = attn[(size_t)b * Ssz + sc0 + tid];
  __syncthreads();
  const float* base = enc + ((size_t)b * Ssz + sc0) * ENCsz;
  float a0 = 0.f, a1 = 0.f;
  for (int s = 0; s < 128; ++s) {
    float w = aw[s];
    a0 = fmaf(w, base[(size_t)s * ENCsz + tid], a0);
    a1 = fmaf(w, base[(size_t)s * ENCsz + tid + 256], a1);
  }
  atomicAdd(&context[(size_t)b * ENCsz + tid], a0);
  atomicAdd(&context[(size_t)b * ENCsz + tid + 256], a1);
}

// ---------------- LSTM cell pointwise + feature packing ----------------
__global__ __launch_bounds__(256)
void lstm_cell_pointwise(const float* __restrict__ gates,
                         const float* __restrict__ b_ih, const float* __restrict__ b_hh,
                         const float* __restrict__ cell, const float* __restrict__ context,
                         const float* __restrict__ embedded,
                         float* __restrict__ h_out, float* __restrict__ c_out,
                         float* __restrict__ feat)
{
  int t = blockIdx.x * 256 + threadIdx.x;   // 128*512
  int b = t >> 9, h = t & 511;
  const float* g = gates + (size_t)b * (4 * Hsz);
  float gi = g[h]           + b_ih[h]           + b_hh[h];
  float gf = g[h + Hsz]     + b_ih[h + Hsz]     + b_hh[h + Hsz];
  float gg = g[h + 2 * Hsz] + b_ih[h + 2 * Hsz] + b_hh[h + 2 * Hsz];
  float go = g[h + 3 * Hsz] + b_ih[h + 3 * Hsz] + b_hh[h + 3 * Hsz];
  float c_new = sigmoidf_(gf) * cell[t] + sigmoidf_(gi) * tanhf(gg);
  float h_new = sigmoidf_(go) * tanhf(c_new);
  h_out[t] = h_new;
  c_out[t] = c_new;
  float* fb = feat + (size_t)b * (Hsz + ENCsz + Esz);
  fb[h] = h_new;
  fb[Hsz + h] = context[t];
  if (h < Esz) fb[Hsz + ENCsz + h] = embedded[(size_t)b * Esz + h];
}

// =========================== launcher ===========================
extern "C" void kernel_launch(void* const* d_in, const int* in_sizes, int n_in,
                              void* d_out, int out_size, void* d_ws, size_t ws_size,
                              hipStream_t stream)
{
  const int*   ids   = (const int*)  d_in[0];
  const float* dh    = (const float*)d_in[1];   // hidden [1,B,H] -> [B,H]
  const float* dc    = (const float*)d_in[2];   // cell
  const float* enc   = (const float*)d_in[3];   // [B,S,ENC]
  const float* emb   = (const float*)d_in[4];   // [V,E]
  const float* Wa    = (const float*)d_in[5];   // [H, H+ENC]
  const float* ba    = (const float*)d_in[6];
  const float* vvec  = (const float*)d_in[7];   // [1,H]
  const float* W_ih  = (const float*)d_in[8];   // [4H, E+ENC]
  const float* W_hh  = (const float*)d_in[9];   // [4H, H]
  const float* b_ih  = (const float*)d_in[10];
  const float* b_hh  = (const float*)d_in[11];
  const float* Wo    = (const float*)d_in[12];  // [V, H+ENC+E]
  const float* bo    = (const float*)d_in[13];

  float* ws = (float*)d_ws;
  float* embedded = ws;                      // 128*256
  float* dec_part = embedded + Bsz * Esz;    // 128*512
  float* scores   = dec_part + Bsz * Hsz;    // 128*1024
  float* context  = scores   + Bsz * Ssz;    // 128*512
  float* gates    = context  + Bsz * ENCsz;  // 128*2048
  float* feat     = gates    + Bsz * 4 * Hsz;// 128*1280

  float* pred  = (float*)d_out;
  float* h_out = pred + (size_t)Bsz * Vsz;
  float* c_out = h_out + Bsz * Hsz;

  // 1) embedding gather + zero score accumulator
  embed_and_zero<<<(Bsz * Ssz) / 256, 256, 0, stream>>>(ids, emb, embedded, scores);

  // 2) dec_part = dh @ Wa[:, :H]^T    (M=128,N=512,K=512)
  gemm_bf16_wmma<<<dim3(1, Hsz / BN), 256, 0, stream>>>(
      dh, Hsz, Wa, Hsz + ENCsz, dec_part, Hsz, nullptr, Bsz, Hsz, Hsz, 0);

  // 3) fused energy GEMM + tanh + v-dot -> scores  (never materializes [B,S,H])
  attn_scores_wmma<<<dim3(Ssz / BM, Hsz / BN, Bsz), 256, 0, stream>>>(
      enc, Wa, ba, vvec, dec_part, scores);

  // 4) softmax over S, zero context accumulator
  softmax_zero_ctx<<<Bsz, 256, 0, stream>>>(scores, context);

  // 5) context = attn . encoder_outputs
  context_accum<<<dim3(Ssz / 128, Bsz), 256, 0, stream>>>(enc, scores, context);

  // 6) gates = embedded @ W_ih[:, :E]^T  (+= context part, += dh @ W_hh^T)
  gemm_bf16_wmma<<<dim3(1, (4 * Hsz) / BN), 256, 0, stream>>>(
      embedded, Esz, W_ih, Esz + ENCsz, gates, 4 * Hsz, nullptr, Bsz, 4 * Hsz, Esz, 0);
  gemm_bf16_wmma<<<dim3(1, (4 * Hsz) / BN), 256, 0, stream>>>(
      context, ENCsz, W_ih + Esz, Esz + ENCsz, gates, 4 * Hsz, nullptr, Bsz, 4 * Hsz, ENCsz, 1);
  gemm_bf16_wmma<<<dim3(1, (4 * Hsz) / BN), 256, 0, stream>>>(
      dh, Hsz, W_hh, Hsz, gates, 4 * Hsz, nullptr, Bsz, 4 * Hsz, Hsz, 1);

  // 7) LSTM pointwise -> h_new, c_new, feat = [h_new | context | embedded]
  lstm_cell_pointwise<<<(Bsz * Hsz) / 256, 256, 0, stream>>>(
      gates, b_ih, b_hh, dc, context, embedded, h_out, c_out, feat);

  // 8) prediction = feat @ Wo^T + bo   (M=128, N=50257, K=1280) -> straight into d_out
  const int KF = Hsz + ENCsz + Esz; // 1280
  gemm_bf16_wmma<<<dim3(1, (Vsz + BN - 1) / BN), 256, 0, stream>>>(
      feat, KF, Wo, KF, pred, Vsz, bo, Bsz, Vsz, KF, 0);
}
